// QILMTransformerBlock_39651138077138
// MI455X (gfx1250) — compile-verified, matrix operand order checked by reading, not run on hardware
//
#include <hip/hip_runtime.h>

// ---------------------------------------------------------------------------
// Problem constants (match reference)
// ---------------------------------------------------------------------------
static constexpr int Bc_ = 4;
static constexpr int Tc_ = 1024;
static constexpr int Dc_ = 1024;
static constexpr int Hc_ = 16;
static constexpr int FFc_ = 4096;
static constexpr int Mrows = Bc_ * Tc_;          // 4096 rows of activations

// ---------------------------------------------------------------------------
// Types for WMMA (CDNA5 gfx1250, wave32)
// ---------------------------------------------------------------------------
typedef __bf16 v16bf  __attribute__((ext_vector_type(16)));
typedef __bf16 bf16x8 __attribute__((ext_vector_type(8)));
typedef float  v8f    __attribute__((ext_vector_type(8)));

__device__ __forceinline__ v8f wmma_bf16(v16bf a, v16bf b, v8f c) {
  // D = A(16x32 bf16) * B(32x16 bf16) + C(16x16 f32)
  return __builtin_amdgcn_wmma_f32_16x16x32_bf16(
      /*neg_a=*/false, a, /*neg_b=*/false, b,
      /*c_mod=*/(short)0, c, /*reuse_a=*/false, /*reuse_b=*/false);
}

// Load a 16x32 bf16 A-fragment (or a B-fragment from an N-major [n][k]
// buffer) following the ISA 7.12.2 layout:
//   lane<16 : row = lane,    K chunks {0..7} and {16..23}
//   lane>=16: row = lane-16, K chunks {8..15} and {24..31}
__device__ __forceinline__ v16bf frag_ld(const __bf16* base, int stride, int lane) {
  const int r  = lane & 15;
  const int k0 = (lane >> 4) << 3;      // 0 or 8
  const __bf16* p = base + r * stride + k0;
  bf16x8 lo = *(const bf16x8*)(p);
  bf16x8 hi = *(const bf16x8*)(p + 16);
  return __builtin_shufflevector(lo, hi, 0,1,2,3,4,5,6,7,8,9,10,11,12,13,14,15);
}

__device__ __forceinline__ v8f v8f_zero() {
  v8f z = {0.f,0.f,0.f,0.f,0.f,0.f,0.f,0.f};
  return z;
}

// ---------------------------------------------------------------------------
// Tensor Data Mover: 2D tile (bf16) global -> LDS, with hardware LDS padding.
// D# packing per CDNA5 ISA 8.3/8.4 (group0: count/lds/global/type,
// group1: data_size, pad, tensor dims, tile dims, dim0 stride).
// This toolchain exposes the 6-arg builtin:
//   (v4u g0, v8i g1, v4i g2, v4i g3, v8i g4, i32 cpol)
// ---------------------------------------------------------------------------
#if defined(__has_builtin)
#if __has_builtin(__builtin_amdgcn_tensor_load_to_lds) && \
    __has_builtin(__builtin_amdgcn_s_wait_tensorcnt)
#define USE_TDM 1
#endif
#endif
#ifndef USE_TDM
#define USE_TDM 0
#endif

#if USE_TDM
typedef unsigned tdm_v4u __attribute__((ext_vector_type(4)));
typedef int      tdm_v8i __attribute__((ext_vector_type(8)));
typedef int      tdm_v4i __attribute__((ext_vector_type(4)));

__device__ __forceinline__ void tdm_load_2d_bf16(
    unsigned lds_off, const __bf16* gptr,
    unsigned tile_w /*elems, contiguous*/, unsigned tile_h /*rows*/,
    unsigned row_stride_elems, unsigned tensor_d0, unsigned tensor_d1,
    unsigned pad_interval, unsigned pad_amount) {
  unsigned long long ga = (unsigned long long)(uintptr_t)gptr;
  tdm_v4u g0;
  g0[0] = 1u;                                            // count=1 (valid)
  g0[1] = lds_off;                                       // lds_addr (bytes)
  g0[2] = (unsigned)(ga & 0xffffffffu);                  // global_addr[31:0]
  g0[3] = (unsigned)((ga >> 32) & 0x01ffffffu)           // global_addr[56:32]
          | (2u << 30);                                  // type = 2 ("image")
  tdm_v8i g1;
  unsigned d0 = (1u << 16);                              // data_size=1 (2B)
  if (pad_amount) d0 |= (1u << 20) | (pad_interval << 22) | (pad_amount << 25);
  g1[0] = (int)d0;
  g1[1] = (int)((tensor_d0 & 0xffffu) << 16);            // abar=0 ; dim0[15:0]
  g1[2] = (int)(((tensor_d0 >> 16) & 0xffffu) | ((tensor_d1 & 0xffffu) << 16));
  g1[3] = (int)(((tensor_d1 >> 16) & 0xffffu) | (tile_w << 16));  // tile_dim0
  g1[4] = (int)(tile_h & 0xffffu);                       // tile_dim1 ; dim2=0
  g1[5] = (int)row_stride_elems;                         // dim0_stride[31:0]
  g1[6] = 0;                                             // stride hi / dim1_stride
  g1[7] = 0;
  tdm_v4i z4 = {0, 0, 0, 0};                             // 2D: groups 2/3 zero
  tdm_v8i z8 = {0, 0, 0, 0, 0, 0, 0, 0};
  __builtin_amdgcn_tensor_load_to_lds(g0, g1, z4, z4, z8, 0);
}
#endif

// ---------------------------------------------------------------------------
// Weight transpose + fp32->bf16 cast: in [K][N] f32 -> out [N][K] bf16.
// 32x32 tiles through LDS; grid (N/32, K/32), block 256 (32x8).
// ---------------------------------------------------------------------------
__global__ __launch_bounds__(256, 1)
void transpose_cast_bf16(const float* __restrict__ in, __bf16* __restrict__ out,
                         int K, int N) {
  __shared__ float tile[32][33];
  const int lx = threadIdx.x & 31;
  const int ly = threadIdx.x >> 5;            // 0..7
  const int n0 = blockIdx.x * 32;
  const int k0 = blockIdx.y * 32;
#pragma unroll
  for (int r = 0; r < 32; r += 8)
    tile[r + ly][lx] = in[(size_t)(k0 + r + ly) * N + n0 + lx];
  __syncthreads();
#pragma unroll
  for (int r = 0; r < 32; r += 8)
    out[(size_t)(n0 + r + ly) * K + k0 + lx] = (__bf16)tile[lx][r + ly];
}

// ---------------------------------------------------------------------------
// Rotary: writes fp32 (for residual) and bf16 (for GEMM inputs)
// grid (D/256, T, B), block 256
// ---------------------------------------------------------------------------
__global__ void rotary_kernel(const float* __restrict__ xr,
                              const float* __restrict__ xi,
                              float* __restrict__ xrf, float* __restrict__ xif,
                              __bf16* __restrict__ xrb, __bf16* __restrict__ xib) {
  const int half = Dc_ / 2;
  int c = blockIdx.x * 256 + threadIdx.x;
  int t = blockIdx.y, b = blockIdx.z;
  size_t o = ((size_t)b * Tc_ + t) * Dc_ + c;
  float r = xr[o], im = xi[o];
  if (c < half) {
    float freq = __expf(-9.210340371976184f * (float)c / (float)half);
    float ang  = (float)t * freq;
    float sn, cs;
    __sincosf(ang, &sn, &cs);
    float r1 = r * cs - im * sn;
    float i1 = r * sn + im * cs;
    r = r1; im = i1;
  }
  xrf[o] = r;  xif[o] = im;
  xrb[o] = (__bf16)r;  xib[o] = (__bf16)im;
}

// ---------------------------------------------------------------------------
// Complex GEMM with bias (+ optional exact GELU on the real part):
//   real = Ar@Wr - Ai@Wi + (br - bi)
//   imag = Ai@Wr + Ar@Wi + (br + bi)
// A: [M,K] bf16 row-major, Wt: [N,K] bf16 (pre-transposed) row-major.
// Block tile 64x128, K-step 32; 8 waves, each a 32x32 sub-tile (2x2 WMMA
// tiles x 4 accumulators P,Q,R,S). Tiles staged to LDS by the TDM (wave 0)
// with hardware padding to a 48-element row stride; VMEM fallback otherwise.
// ---------------------------------------------------------------------------
__global__ __launch_bounds__(256, 1)
void cgemm_bf16(const __bf16* __restrict__ Ar, const __bf16* __restrict__ Ai,
                const __bf16* __restrict__ Wtr, const __bf16* __restrict__ Wti,
                const float* __restrict__ br, const float* __restrict__ bi,
                float* outRf, float* outIf, __bf16* outRb, __bf16* outIb,
                int M, int N, int K, int act) {
  constexpr int BM = 64, BN = 128, BK = 32;
  constexpr int LD = 48;                      // padded rows: 96B (16B aligned)
  __shared__ __bf16 sAr[BM * LD];
  __shared__ __bf16 sAi[BM * LD];
  __shared__ __bf16 sBr[BN * LD];             // Wt tile, [n][k]
  __shared__ __bf16 sBi[BN * LD];

  const int tid  = threadIdx.x;
  const int lane = tid & 31;
  const int wave = tid >> 5;
  const int m0 = blockIdx.y * BM;
  const int n0 = blockIdx.x * BN;
  const int wm = (wave >> 2) * 32;            // 2 waves along M
  const int wn = (wave & 3) * 32;             // 4 waves along N

  v8f accP[2][2], accQ[2][2], accR[2][2], accS[2][2];
#pragma unroll
  for (int mi = 0; mi < 2; ++mi)
#pragma unroll
    for (int ni = 0; ni < 2; ++ni) {
      accP[mi][ni] = v8f_zero(); accQ[mi][ni] = v8f_zero();
      accR[mi][ni] = v8f_zero(); accS[mi][ni] = v8f_zero();
    }

  for (int k = 0; k < K; k += BK) {
    __syncthreads();                          // previous tile fully consumed
#if USE_TDM
    if (tid < 32) {                           // one wave drives the TDM
      // pad: 16 DWORDs (one 32-elem row) then +8 DWORDs -> LDS stride 48
      tdm_load_2d_bf16((unsigned)(uintptr_t)sAr, Ar + (size_t)m0 * K + k,
                       BK, BM, (unsigned)K, (unsigned)K, (unsigned)M, 3u, 7u);
      tdm_load_2d_bf16((unsigned)(uintptr_t)sAi, Ai + (size_t)m0 * K + k,
                       BK, BM, (unsigned)K, (unsigned)K, (unsigned)M, 3u, 7u);
      tdm_load_2d_bf16((unsigned)(uintptr_t)sBr, Wtr + (size_t)n0 * K + k,
                       BK, BN, (unsigned)K, (unsigned)K, (unsigned)N, 3u, 7u);
      tdm_load_2d_bf16((unsigned)(uintptr_t)sBi, Wti + (size_t)n0 * K + k,
                       BK, BN, (unsigned)K, (unsigned)K, (unsigned)N, 3u, 7u);
      __builtin_amdgcn_s_wait_tensorcnt(0);
    }
#else
    {                                         // A: 64 rows x 4 chunks of 8
      int arow = tid >> 2, acol = (tid & 3) << 3;
      bf16x8 va_r = *(const bf16x8*)(Ar + (size_t)(m0 + arow) * K + k + acol);
      bf16x8 va_i = *(const bf16x8*)(Ai + (size_t)(m0 + arow) * K + k + acol);
      *(bf16x8*)(sAr + arow * LD + acol) = va_r;
      *(bf16x8*)(sAi + arow * LD + acol) = va_i;
    }
#pragma unroll
    for (int c = tid; c < 512; c += 256) {    // B: 128 rows x 4 chunks of 8
      int row = c >> 2, kc = (c & 3) << 3;
      bf16x8 wr8 = *(const bf16x8*)(Wtr + (size_t)(n0 + row) * K + k + kc);
      bf16x8 wi8 = *(const bf16x8*)(Wti + (size_t)(n0 + row) * K + k + kc);
      *(bf16x8*)(sBr + row * LD + kc) = wr8;
      *(bf16x8*)(sBi + row * LD + kc) = wi8;
    }
#endif
    __syncthreads();

    v16bf fAr[2], fAi[2], fBr[2], fBi[2];
#pragma unroll
    for (int mi = 0; mi < 2; ++mi) {
      fAr[mi] = frag_ld(sAr + (wm + mi * 16) * LD, LD, lane);
      fAi[mi] = frag_ld(sAi + (wm + mi * 16) * LD, LD, lane);
    }
#pragma unroll
    for (int ni = 0; ni < 2; ++ni) {
      fBr[ni] = frag_ld(sBr + (wn + ni * 16) * LD, LD, lane);
      fBi[ni] = frag_ld(sBi + (wn + ni * 16) * LD, LD, lane);
    }
#pragma unroll
    for (int mi = 0; mi < 2; ++mi)
#pragma unroll
      for (int ni = 0; ni < 2; ++ni) {
        accP[mi][ni] = wmma_bf16(fAr[mi], fBr[ni], accP[mi][ni]);
        accQ[mi][ni] = wmma_bf16(fAi[mi], fBi[ni], accQ[mi][ni]);
        accR[mi][ni] = wmma_bf16(fAi[mi], fBr[ni], accR[mi][ni]);
        accS[mi][ni] = wmma_bf16(fAr[mi], fBi[ni], accS[mi][ni]);
      }
  }

  // ---- epilogue: C/D layout = lane%16 -> N col, VGPR i + 8*(lane/16) -> M row
#pragma unroll
  for (int ni = 0; ni < 2; ++ni) {
    int col = n0 + wn + ni * 16 + (lane & 15);
    float bre = br[col], bim = bi[col];
    float biasR = bre - bim;
    float biasI = bre + bim;
#pragma unroll
    for (int mi = 0; mi < 2; ++mi) {
#pragma unroll
      for (int i = 0; i < 8; ++i) {
        int row = m0 + wm + mi * 16 + i + ((lane >> 4) << 3);
        float re = accP[mi][ni][i] - accQ[mi][ni][i] + biasR;
        float im = accR[mi][ni][i] + accS[mi][ni][i] + biasI;
        if (act == 1) re = 0.5f * re * (1.0f + erff(re * 0.70710678118f));
        size_t o = (size_t)row * N + col;
        if (outRf) outRf[o] = re;
        if (outIf) outIf[o] = im;
        if (outRb) outRb[o] = (__bf16)re;
        if (outIb) outIb[o] = (__bf16)im;
      }
    }
  }
}

// ---------------------------------------------------------------------------
// Flash attention over complex q/k (real scores) with real probs@v.
//   scores = (qr.kr + qi.ki) / 8 ; probs = softmax ; o = probs @ v (r and i)
// q/k/v bf16 [B,T,D]; head h = columns [h*64, h*64+64).
// grid (T/64, H, B); 128 threads = 4 waves; each wave owns 16 query rows.
// KV processed 32 rows at a time; K tiles staged by the TDM (stride 72 via
// pad), V transposed in software for the P@V B-fragments.
// ---------------------------------------------------------------------------
__global__ __launch_bounds__(128, 1)
void attn_fwd(const __bf16* __restrict__ qr, const __bf16* __restrict__ qi,
              const __bf16* __restrict__ kr, const __bf16* __restrict__ ki,
              const __bf16* __restrict__ vr, const __bf16* __restrict__ vi,
              __bf16* __restrict__ outr, __bf16* __restrict__ outi) {
  constexpr int dh  = 64;
  constexpr int LDK = 72;   // 144B rows (16B aligned)
  constexpr int LDV = 48;   // 96B rows
  constexpr int LDP = 48;
  __shared__ __bf16 sKr[32 * LDK];
  __shared__ __bf16 sKi[32 * LDK];
  __shared__ __bf16 sVtr[dh * LDV];           // V transposed: [d][kv]
  __shared__ __bf16 sVti[dh * LDV];
  __shared__ __bf16 sP[4][16 * LDP];          // per-wave P staging

  const int tid  = threadIdx.x;
  const int lane = tid & 31;
  const int wave = tid >> 5;
  const int b = blockIdx.z, h = blockIdx.y, mt = blockIdx.x;
  const int qrow  = mt * 64 + wave * 16;
  const size_t baseQ = ((size_t)b * Tc_ + qrow) * Dc_ + h * dh;

  v16bf fqr[2], fqi[2];
  fqr[0] = frag_ld(qr + baseQ,      Dc_, lane);
  fqr[1] = frag_ld(qr + baseQ + 32, Dc_, lane);
  fqi[0] = frag_ld(qi + baseQ,      Dc_, lane);
  fqi[1] = frag_ld(qi + baseQ + 32, Dc_, lane);

  v8f Or[4], Oi[4];
#pragma unroll
  for (int dt = 0; dt < 4; ++dt) { Or[dt] = v8f_zero(); Oi[dt] = v8f_zero(); }
  float mrow[8], lrow[8];
#pragma unroll
  for (int i = 0; i < 8; ++i) { mrow[i] = -1e30f; lrow[i] = 0.f; }

  __bf16* myP = sP[wave];

  for (int j = 0; j < Tc_ / 32; ++j) {
    const size_t baseKV = ((size_t)b * Tc_ + j * 32) * Dc_ + h * dh;
    __syncthreads();                          // previous K/V fully consumed
#if USE_TDM
    if (tid < 32) {
      // 64-elem rows (32 DWORDs) then +4 DWORDs pad -> LDS stride 72
      tdm_load_2d_bf16((unsigned)(uintptr_t)sKr, kr + baseKV,
                       dh, 32, (unsigned)Dc_, (unsigned)Dc_,
                       (unsigned)(Bc_ * Tc_), 4u, 3u);
      tdm_load_2d_bf16((unsigned)(uintptr_t)sKi, ki + baseKV,
                       dh, 32, (unsigned)Dc_, (unsigned)Dc_,
                       (unsigned)(Bc_ * Tc_), 4u, 3u);
    }
#endif
#pragma unroll
    for (int c = tid; c < 256; c += 128) {    // V (+K fallback): 32r x 8 chunks
      int row = c >> 3;
      int dc  = (c & 7) << 3;
      size_t g = baseKV + (size_t)row * Dc_ + dc;
      bf16x8 xvr = *(const bf16x8*)(vr + g);
      bf16x8 xvi = *(const bf16x8*)(vi + g);
#if !USE_TDM
      bf16x8 xkr = *(const bf16x8*)(kr + g);
      bf16x8 xki = *(const bf16x8*)(ki + g);
      *(bf16x8*)(sKr + row * LDK + dc) = xkr;
      *(bf16x8*)(sKi + row * LDK + dc) = xki;
#endif
#pragma unroll
      for (int e = 0; e < 8; ++e) {
        sVtr[(dc + e) * LDV + row] = xvr[e];
        sVti[(dc + e) * LDV + row] = xvi[e];
      }
    }
#if USE_TDM
    if (tid < 32) __builtin_amdgcn_s_wait_tensorcnt(0);
#endif
    __syncthreads();

    // ---- scores: S[16 x 32] = qr.kr^T + qi.ki^T (k rows ARE B-fragments) --
    v8f S0 = v8f_zero(), S1 = v8f_zero();
    v16bf fk;
    fk = frag_ld(sKr,                 LDK, lane); S0 = wmma_bf16(fqr[0], fk, S0);
    fk = frag_ld(sKr + 32,            LDK, lane); S0 = wmma_bf16(fqr[1], fk, S0);
    fk = frag_ld(sKi,                 LDK, lane); S0 = wmma_bf16(fqi[0], fk, S0);
    fk = frag_ld(sKi + 32,            LDK, lane); S0 = wmma_bf16(fqi[1], fk, S0);
    fk = frag_ld(sKr + 16 * LDK,      LDK, lane); S1 = wmma_bf16(fqr[0], fk, S1);
    fk = frag_ld(sKr + 16 * LDK + 32, LDK, lane); S1 = wmma_bf16(fqr[1], fk, S1);
    fk = frag_ld(sKi + 16 * LDK,      LDK, lane); S1 = wmma_bf16(fqi[0], fk, S1);
    fk = frag_ld(sKi + 16 * LDK + 32, LDK, lane); S1 = wmma_bf16(fqi[1], fk, S1);

    // ---- online softmax (row reductions across 16-lane halves) ----
    const int prow0 = (lane >> 4) << 3;
#pragma unroll
    for (int i = 0; i < 8; ++i) {
      float s0 = S0[i] * 0.125f;
      float s1 = S1[i] * 0.125f;
      float mx = fmaxf(s0, s1);
#pragma unroll
      for (int off = 1; off < 16; off <<= 1)
        mx = fmaxf(mx, __shfl_xor(mx, off, 32));
      float nm    = fmaxf(mrow[i], mx);
      float alpha = __expf(mrow[i] - nm);
      float p0 = __expf(s0 - nm);
      float p1 = __expf(s1 - nm);
      float rs = p0 + p1;
#pragma unroll
      for (int off = 1; off < 16; off <<= 1)
        rs += __shfl_xor(rs, off, 32);
      lrow[i] = lrow[i] * alpha + rs;
      mrow[i] = nm;
#pragma unroll
      for (int dt = 0; dt < 4; ++dt) { Or[dt][i] *= alpha; Oi[dt][i] *= alpha; }
      int pr = i + prow0;
      myP[pr * LDP + (lane & 15)]      = (__bf16)p0;
      myP[pr * LDP + 16 + (lane & 15)] = (__bf16)p1;
    }

    // ---- O += P @ V  (P as A fragment; V^T rows are B fragments) ----
    v16bf fP = frag_ld(myP, LDP, lane);
#pragma unroll
    for (int dt = 0; dt < 4; ++dt) {
      v16bf fvr = frag_ld(sVtr + dt * 16 * LDV, LDV, lane);
      v16bf fvi = frag_ld(sVti + dt * 16 * LDV, LDV, lane);
      Or[dt] = wmma_bf16(fP, fvr, Or[dt]);
      Oi[dt] = wmma_bf16(fP, fvi, Oi[dt]);
    }
  }

  // ---- normalize and store bf16 ----
#pragma unroll
  for (int i = 0; i < 8; ++i) {
    float inv = 1.0f / lrow[i];
    int r = i + ((lane >> 4) << 3);
#pragma unroll
    for (int dt = 0; dt < 4; ++dt) {
      size_t o = baseQ + (size_t)r * Dc_ + dt * 16 + (lane & 15);
      outr[o] = (__bf16)(Or[dt][i] * inv);
      outi[o] = (__bf16)(Oi[dt][i] * inv);
    }
  }
}

// ---------------------------------------------------------------------------
// Residual + LayerNorm: out = LN(x + a) * g + b ; fp32 and/or bf16 outputs.
// One block (256 threads) per 1024-wide row.
// ---------------------------------------------------------------------------
__global__ __launch_bounds__(256, 1)
void resid_layernorm(const float* __restrict__ x, const float* __restrict__ a,
                     const float* __restrict__ g, const float* __restrict__ bb,
                     float* outF, __bf16* outB) {
  const int row = blockIdx.x;
  const int tid = threadIdx.x;
  const float* px = x + (size_t)row * Dc_;
  const float* pa = a ? a + (size_t)row * Dc_ : nullptr;
  float v[4];
  float sum = 0.f, sq = 0.f;
#pragma unroll
  for (int i = 0; i < 4; ++i) {
    int c = tid + i * 256;
    float t = px[c] + (pa ? pa[c] : 0.0f);
    v[i] = t; sum += t; sq += t * t;
  }
  __shared__ float sbuf[16];
#pragma unroll
  for (int off = 16; off >= 1; off >>= 1) {
    sum += __shfl_xor(sum, off, 32);
    sq  += __shfl_xor(sq,  off, 32);
  }
  const int lane = tid & 31, wave = tid >> 5;
  if (lane == 0) { sbuf[wave] = sum; sbuf[8 + wave] = sq; }
  __syncthreads();
  if (wave == 0) {
    float s = (lane < 8) ? sbuf[lane] : 0.f;
    float q = (lane < 8) ? sbuf[8 + lane] : 0.f;
#pragma unroll
    for (int off = 4; off >= 1; off >>= 1) {
      s += __shfl_xor(s, off, 32);
      q += __shfl_xor(q, off, 32);
    }
    if (lane == 0) { sbuf[0] = s; sbuf[1] = q; }
  }
  __syncthreads();
  float mean = sbuf[0] * (1.0f / Dc_);
  float var  = sbuf[1] * (1.0f / Dc_) - mean * mean;
  float rstd = rsqrtf(var + 1e-5f);
#pragma unroll
  for (int i = 0; i < 4; ++i) {
    int c = tid + i * 256;
    float t = (v[i] - mean) * rstd * g[c] + bb[c];
    size_t o = (size_t)row * Dc_ + c;
    if (outF) outF[o] = t;
    if (outB) outB[o] = (__bf16)t;
  }
}

// ---------------------------------------------------------------------------
// Orchestration
// ---------------------------------------------------------------------------
extern "C" void kernel_launch(void* const* d_in, const int* in_sizes, int n_in,
                              void* d_out, int out_size, void* d_ws, size_t ws_size,
                              hipStream_t stream) {
  (void)in_sizes; (void)n_in; (void)out_size; (void)ws_size;
  const float* x_real = (const float*)d_in[0];
  const float* x_imag = (const float*)d_in[1];
  const float* W[12]; const float* bias[12];
  for (int p = 0; p < 6; ++p) {                     // q,k,v,o,ff1,ff2
    W[2 * p + 0]    = (const float*)d_in[2 + 4 * p + 0];   // Wr
    W[2 * p + 1]    = (const float*)d_in[2 + 4 * p + 1];   // Wi
    bias[2 * p + 0] = (const float*)d_in[2 + 4 * p + 2];   // br
    bias[2 * p + 1] = (const float*)d_in[2 + 4 * p + 3];   // bi
  }
  const float* n1r_g = (const float*)d_in[26]; const float* n1r_b = (const float*)d_in[27];
  const float* n1i_g = (const float*)d_in[28]; const float* n1i_b = (const float*)d_in[29];
  const float* n2r_g = (const float*)d_in[30]; const float* n2r_b = (const float*)d_in[31];
  const float* n2i_g = (const float*)d_in[32]; const float* n2i_b = (const float*)d_in[33];

  char* ws = (char*)d_ws;
  size_t off = 0;
  auto take = [&](size_t bytes) -> char* {
    char* p = ws + off;
    off += (bytes + 255) & ~(size_t)255;
    return p;
  };
  const size_t DD  = (size_t)Dc_ * Dc_;
  const size_t DFF = (size_t)Dc_ * FFc_;
  const size_t AD  = (size_t)Mrows * Dc_;
  const size_t AFF = (size_t)Mrows * FFc_;

  // Transposed bf16 weights: wt[i] is [N][K] for weight i
  __bf16* wt[12];
  int wK[12], wN[12];
  for (int i = 0; i < 8; ++i)  { wt[i] = (__bf16*)take(DD * 2);  wK[i] = Dc_;  wN[i] = Dc_; }
  wt[8]  = (__bf16*)take(DFF * 2); wK[8]  = Dc_;  wN[8]  = FFc_;   // ff1 Wr
  wt[9]  = (__bf16*)take(DFF * 2); wK[9]  = Dc_;  wN[9]  = FFc_;   // ff1 Wi
  wt[10] = (__bf16*)take(DFF * 2); wK[10] = FFc_; wN[10] = Dc_;    // ff2 Wr
  wt[11] = (__bf16*)take(DFF * 2); wK[11] = FFc_; wN[11] = Dc_;    // ff2 Wi

  float*  xrot_rf = (float*)take(AD * 4);
  float*  xrot_if = (float*)take(AD * 4);
  __bf16* xrot_rb = (__bf16*)take(AD * 2);
  __bf16* xrot_ib = (__bf16*)take(AD * 2);
  __bf16* q_r = (__bf16*)take(AD * 2); __bf16* q_i = (__bf16*)take(AD * 2);
  __bf16* k_r = (__bf16*)take(AD * 2); __bf16* k_i = (__bf16*)take(AD * 2);
  __bf16* v_r = (__bf16*)take(AD * 2); __bf16* v_i = (__bf16*)take(AD * 2);
  __bf16* at_r = (__bf16*)take(AD * 2); __bf16* at_i = (__bf16*)take(AD * 2);
  float*  a_r = (float*)take(AD * 4);  float*  a_i = (float*)take(AD * 4);
  float*  n1_rf = (float*)take(AD * 4); float* n1_if = (float*)take(AD * 4);
  __bf16* n1_rb = (__bf16*)take(AD * 2); __bf16* n1_ib = (__bf16*)take(AD * 2);
  __bf16* f_r = (__bf16*)take(AFF * 2); __bf16* f_i = (__bf16*)take(AFF * 2);
  float*  g_r = (float*)take(AD * 4);  float*  g_i = (float*)take(AD * 4);

  // ---- 1) transpose+cast weights to bf16 [N][K] ----
  for (int i = 0; i < 12; ++i) {
    transpose_cast_bf16<<<dim3(wN[i] / 32, wK[i] / 32), dim3(256), 0, stream>>>(
        W[i], wt[i], wK[i], wN[i]);
  }

  // ---- 2) rotary ----
  rotary_kernel<<<dim3(Dc_ / 256, Tc_, Bc_), dim3(256), 0, stream>>>(
      x_real, x_imag, xrot_rf, xrot_if, xrot_rb, xrot_ib);

  // ---- 3) q/k/v projections (bf16 outputs only) ----
  dim3 gProj(Dc_ / 128, Mrows / 64), blk(256);
  cgemm_bf16<<<gProj, blk, 0, stream>>>(xrot_rb, xrot_ib, wt[0], wt[1],
      bias[0], bias[1], nullptr, nullptr, q_r, q_i, Mrows, Dc_, Dc_, 0);
  cgemm_bf16<<<gProj, blk, 0, stream>>>(xrot_rb, xrot_ib, wt[2], wt[3],
      bias[2], bias[3], nullptr, nullptr, k_r, k_i, Mrows, Dc_, Dc_, 0);
  cgemm_bf16<<<gProj, blk, 0, stream>>>(xrot_rb, xrot_ib, wt[4], wt[5],
      bias[4], bias[5], nullptr, nullptr, v_r, v_i, Mrows, Dc_, Dc_, 0);

  // ---- 4) attention ----
  attn_fwd<<<dim3(Tc_ / 64, Hc_, Bc_), dim3(128), 0, stream>>>(
      q_r, q_i, k_r, k_i, v_r, v_i, at_r, at_i);

  // ---- 5) output projection (fp32 outputs for residual) ----
  cgemm_bf16<<<gProj, blk, 0, stream>>>(at_r, at_i, wt[6], wt[7],
      bias[6], bias[7], a_r, a_i, nullptr, nullptr, Mrows, Dc_, Dc_, 0);

  // ---- 6) n1 = LN(xrot + attn_out) ----
  resid_layernorm<<<dim3(Mrows), dim3(256), 0, stream>>>(
      xrot_rf, a_r, n1r_g, n1r_b, n1_rf, n1_rb);
  resid_layernorm<<<dim3(Mrows), dim3(256), 0, stream>>>(
      xrot_if, a_i, n1i_g, n1i_b, n1_if, n1_ib);

  // ---- 7) ff1 (+GELU on real), bf16 outputs ----
  cgemm_bf16<<<dim3(FFc_ / 128, Mrows / 64), blk, 0, stream>>>(
      n1_rb, n1_ib, wt[8], wt[9], bias[8], bias[9],
      nullptr, nullptr, f_r, f_i, Mrows, FFc_, Dc_, 1);

  // ---- 8) ff2, fp32 outputs ----
  cgemm_bf16<<<dim3(Dc_ / 128, Mrows / 64), blk, 0, stream>>>(
      f_r, f_i, wt[10], wt[11], bias[10], bias[11],
      g_r, g_i, nullptr, nullptr, Mrows, Dc_, FFc_, 0);

  // ---- 9) final layernorms straight into d_out (outr, then outi) ----
  float* outr = (float*)d_out;
  float* outi = outr + AD;
  resid_layernorm<<<dim3(Mrows), dim3(256), 0, stream>>>(
      n1_rf, g_r, n2r_g, n2r_b, outr, nullptr);
  resid_layernorm<<<dim3(Mrows), dim3(256), 0, stream>>>(
      n1_if, g_i, n2i_g, n2i_b, outi, nullptr);
}
